// GeRN_32744830664990
// MI455X (gfx1250) — matrix operand order repaired and checked by reading, not compiled, not run on hardware
//
#include <hip/hip_runtime.h>
#include <hip/hip_bf16.h>
#include <math.h>

typedef __bf16 bf16_t;
typedef bf16_t v16bf __attribute__((ext_vector_type(16)));
typedef float  v8f   __attribute__((ext_vector_type(8)));

#define NRDIM 256
#define NQDIM 64
#define NHDIM 64
#define BATCH 8192
#define TROWS 16

#if defined(__has_builtin)
#if __has_builtin(__builtin_amdgcn_sched_barrier)
#define SCHED_FENCE() __builtin_amdgcn_sched_barrier(0)
#endif
#endif
#ifndef SCHED_FENCE
#define SCHED_FENCE()
#endif

// ---------------- LDS layout (bytes, per wave; 1 wave per WG) ----------------
// Contiguous per-GEMM A buffers (bf16, 16 rows, stride = padded K):
//  A_ij : [cnd 0..255 | qj 256..319 | hgj 320..383 | hij 384..447]  stride 448
//  A_id : [cnd | qd | hgd | hid]                                    stride 448
//  A_gj : [cnd | zj 256..319 | hgj 320..383]                        stride 384
//  A_gd : [cnd | zd 256..319 | qv 320..351(pad) | hgd 352..415]     stride 416
//  A_dj : [ugb 0..63 | hgj 64..127]                                 stride 128
//  A_dd : [ugb | hgd]                                               stride 128
#define OF_AIJ 0
#define OF_AID 14336
#define OF_AGJ 28672
#define OF_AGD 40960
#define OF_ADJ 54272
#define OF_ADD 58368
#define OF_CIJ 62464   // f32 16x64 cell states
#define OF_CID 66560
#define OF_CGJ 70656
#define OF_CGD 74752
#define OF_UGJ 78848   // f32 16x64 canvases
#define OF_UGD 82944
#define PER_WAVE 87040

// ---------------- packed-weight workspace offsets (bf16 elems) ----------------
#define WS_IJ  0u         // 14 Kt x 16 Nt
#define WS_ID  114688u
#define WS_GJ  229376u    // 12 x 16
#define WS_GD  327680u    // 13 x 16 (remapped)
#define WS_POJ 434176u    // 2 x 8
#define WS_POD 442368u
#define WS_PRJ 450560u
#define WS_PRD 458752u
#define WS_DJ  466944u    // 4 x 4
#define WS_DD  475136u

// =====================================================================
// Prep: pack fp32 weights -> bf16 WMMA B-fragment tiles.
// Tile (kt,nt) stored at (nt*numKt+kt)*512; lane L owns 16 contiguous bf16:
//   value j of lane L = W[kt*32 + (j<8 ? (L/16)*8 + j : 16 + (L/16)*8 + j-8)][nt*16 + (L%16)]
// remapGd handles W_gd's [cnd(256)|z(64)|qv(4)|h(64)] -> padded K=416 layout.
// =====================================================================
__global__ void gern_prep_w(const float* __restrict__ W, int fanIn, int fanOut,
                            int numKt, int numNt, bf16_t* __restrict__ dst,
                            int remapGd) {
  int flat = blockIdx.x * blockDim.x + threadIdx.x;
  int total = numKt * numNt * 512;
  if (flat >= total) return;
  int tile = flat >> 9;
  int lane = (flat >> 4) & 31;
  int j    = flat & 15;
  int nt = tile / numKt;
  int kt = tile - nt * numKt;
  int kLocal = (j < 8) ? (((lane >> 4) << 3) + j) : (16 + ((lane >> 4) << 3) + (j - 8));
  int nLocal = lane & 15;
  int kp = kt * 32 + kLocal;
  int n  = nt * 16 + nLocal;
  int row;
  if (remapGd) {               // padded K: [0,324)=rows 0..323, [324,352)=pad, [352,416)=rows 324..387
    if (kp < 324)      row = kp;
    else if (kp < 352) row = -1;
    else               row = kp - 28;
  } else {
    row = kp;
  }
  float v = 0.0f;
  if (row >= 0 && row < fanIn && n < fanOut) v = W[(size_t)row * fanOut + n];
  dst[(size_t)tile * 512 + lane * 16 + j] = (bf16_t)v;
}

// =====================================================================
// Fragment loaders / math helpers
// =====================================================================
__device__ __forceinline__ v8f zero8() {
  v8f z;
#pragma unroll
  for (int i = 0; i < 8; ++i) z[i] = 0.0f;
  return z;
}

// 16-bit A 16x32 layout: lanes0-15 m=lane, runs {k..k+7, k+16..k+23};
// lanes16-31 m=lane-16, runs shifted by +8.
__device__ __forceinline__ v16bf load_a(const bf16_t* base, int stride, int kOff,
                                        int lane) {
  int m  = lane & 15;
  int kb = kOff + ((lane >> 4) << 3);
  const bf16_t* p = base + m * stride + kb;
  union { uint4 u[2]; v16bf v; } f;
  f.u[0] = *(const uint4*)(p);
  f.u[1] = *(const uint4*)(p + 16);
  return f.v;
}

__device__ __forceinline__ v16bf load_b(const bf16_t* tile, int lane) {
  union { uint4 u[2]; v16bf v; } f;
  const uint4* p = (const uint4*)(tile + (lane << 4));
  f.u[0] = p[0];
  f.u[1] = p[1];
  return f.v;
}

__device__ __forceinline__ v8f wmma_bf16(v16bf a, v16bf b, v8f c) {
  return __builtin_amdgcn_wmma_f32_16x16x32_bf16(false, a, false, b, (short)0, c,
                                                 false, false);
}

__device__ __forceinline__ float sigmoid_fast(float x) {
  return 1.0f / (1.0f + __expf(-x));
}
__device__ __forceinline__ float tanh_fast(float x) {
  float e = __expf(2.0f * x);
  return 1.0f - 2.0f / (e + 1.0f);
}

__device__ __forceinline__ float eps_rand(unsigned which, unsigned t,
                                          unsigned row, unsigned n) {
  unsigned s = (which * 0x9E3779B9u) ^ (t * 0x85EBCA6Bu) ^ (row * 0xC2B2AE35u) ^
               (n * 0x27D4EB2Fu) ^ 0xA511E9B3u;
  float acc = 0.0f;
#pragma unroll
  for (int k = 0; k < 4; ++k) {
    s ^= s << 13; s ^= s >> 17; s ^= s << 5;
    acc += (float)(s >> 8) * 5.9604645e-8f;   // 2^-24
  }
  return (acc - 2.0f) * 1.7320508f;           // CLT approx N(0,1)
}

struct HDst { bf16_t* base; int stride; int colOff; };

// One pass over all K chunks accumulating 8 N-tiles. The K loop is kept
// rolled (unroll(disable)) so CFG back-edges bound scheduler hoisting:
// per iteration 1 A-frag + 8 B-frags in flight -> no spills.
template <int NKT>
__device__ __forceinline__ void gemm_half(const bf16_t* aBase, const bf16_t* w,
                                          int lane, v8f* acc /*8*/) {
#pragma clang loop unroll(disable)
  for (int kt = 0; kt < NKT; ++kt) {
    v16bf a = load_a(aBase, NKT * 32, kt * 32, lane);
    const bf16_t* wtile = w + (size_t)kt * 512;
#pragma unroll
    for (int q = 0; q < 8; ++q) {
      v16bf b = load_b(wtile + (size_t)q * NKT * 512, lane);
      acc[q] = wmma_bf16(a, b, acc[q]);
    }
  }
}

// =====================================================================
// Fused LSTM: [16 x NKT*32] @ W[NKT*32 x 256] -> gates -> h/c update.
// N in two halves of 8 tiles (i|f then g|o); for hidden unit (m,n) the
// i/f/g/o pre-activations live in accIF[q], accIF[q+4], accGO[q], accGO[q+4]
// of the SAME lane (D layout: lane = (m/8)*16 + n%16): register-only gates.
// New h is broadcast (bf16) into up to 3 A-buffer slots for later GEMMs.
// =====================================================================
template <int NKT>
__device__ __forceinline__ void lstm_fused(const bf16_t* aBase,
                                           const bf16_t* w,
                                           const float* __restrict__ bias,
                                           float* c, const HDst* hd, float zo,
                                           int lane) {
  v8f accIF[8];
#pragma unroll
  for (int q = 0; q < 8; ++q) accIF[q] = zero8();
  gemm_half<NKT>(aBase, w, lane, accIF);
  SCHED_FENCE();

  v8f accGO[8];
#pragma unroll
  for (int q = 0; q < 8; ++q) accGO[q] = zero8();
  gemm_half<NKT>(aBase, w + (size_t)8 * NKT * 512, lane, accGO);
  SCHED_FENCE();

  const int col16 = lane & 15;
  const int row0  = (lane >> 4) << 3;
#pragma unroll
  for (int q = 0; q < 4; ++q) {
    int n = q * 16 + col16;
    float bi = bias[n];
    float bf_ = bias[n + 64];
    float bg = bias[n + 128];
    float bo = bias[n + 192];
#pragma unroll
    for (int r = 0; r < 8; ++r) {
      int m = row0 + r;
      int idx = m * 64 + n;
      float si = sigmoid_fast(accIF[q][r] + bi);
      float sf = sigmoid_fast(accIF[q + 4][r] + bf_);
      float tg = tanh_fast(accGO[q][r] + bg);
      float so = sigmoid_fast(accGO[q + 4][r] + bo);
      float cold = c[idx];
      float cnew = sf * cold + si * tg;
      float hnew = so * tanh_fast(cnew);
      float hold = (float)hd[0].base[m * hd[0].stride + hd[0].colOff + n];
      c[idx] = zo * cold + (1.0f - zo) * cnew;
      bf16_t hb = (bf16_t)(zo * hold + (1.0f - zo) * hnew);
#pragma unroll
      for (int d = 0; d < 3; ++d)
        if (hd[d].base)
          hd[d].base[m * hd[d].stride + hd[d].colOff + n] = hb;
    }
  }
  SCHED_FENCE();
}

// Fused Gaussian head: [16x64] @ W[64x128]; mean in acc[q], logv in acc[q+4].
// Reads the h slice directly out of an A buffer (base+colOff, given stride).
// Writes mean/logv slabs to global; optionally samples z into an A buffer.
__device__ __forceinline__ void head_fused(
    const bf16_t* hsrc, int hstride, const bf16_t* w,
    const float* __restrict__ bias, float* __restrict__ outMean,
    float* __restrict__ outLogv, bf16_t* zbase, int zstride, int zcol,
    unsigned which, unsigned t, int rowBase, int lane) {
  v8f acc[8];
#pragma unroll
  for (int q = 0; q < 8; ++q) acc[q] = zero8();
#pragma clang loop unroll(disable)
  for (int kt = 0; kt < 2; ++kt) {
    v16bf a = load_a(hsrc, hstride, kt * 32, lane);
    const bf16_t* wtile = w + (size_t)kt * 512;
#pragma unroll
    for (int q = 0; q < 8; ++q) {
      v16bf b = load_b(wtile + (size_t)q * 2 * 512, lane);
      acc[q] = wmma_bf16(a, b, acc[q]);
    }
  }
  const int col16 = lane & 15;
  const int row0  = (lane >> 4) << 3;
#pragma unroll
  for (int q = 0; q < 4; ++q) {
    int n = q * 16 + col16;
    float bm = bias[n];
    float bl = bias[n + 64];
#pragma unroll
    for (int r = 0; r < 8; ++r) {
      int m = row0 + r;
      float mean = acc[q][r] + bm;
      float logv = acc[q + 4][r] + bl;
      size_t g = (size_t)(rowBase + m) * NHDIM + n;
      outMean[g] = mean;
      outLogv[g] = logv;
      if (zbase) {
        float ev = eps_rand(which, t, (unsigned)(rowBase + m), (unsigned)n);
        zbase[m * zstride + zcol + n] =
            (bf16_t)(mean + __expf(0.5f * logv) * ev);
      }
    }
  }
  SCHED_FENCE();
}

// Fused canvas delta: [16x128] @ W[128x64] -> leaky_relu -> ug += d.
// A buffer holds [ugb | hg]; writes updated ug (f32) and its bf16 copy back.
__device__ __forceinline__ void delta_fused(const bf16_t* aBase,
                                            const bf16_t* w,
                                            const float* __restrict__ bias,
                                            float* ug, bf16_t* ugb, int lane) {
  v8f acc[4];
#pragma unroll
  for (int q = 0; q < 4; ++q) acc[q] = zero8();
#pragma clang loop unroll(disable)
  for (int kt = 0; kt < 4; ++kt) {
    v16bf a = load_a(aBase, 128, kt * 32, lane);
    const bf16_t* wtile = w + (size_t)kt * 512;
#pragma unroll
    for (int q = 0; q < 4; ++q) {
      v16bf b = load_b(wtile + (size_t)q * 4 * 512, lane);
      acc[q] = wmma_bf16(a, b, acc[q]);
    }
  }
  const int col16 = lane & 15;
  const int row0  = (lane >> 4) << 3;
#pragma unroll
  for (int q = 0; q < 4; ++q) {
    int n = q * 16 + col16;
    float bv = bias[n];
#pragma unroll
    for (int r = 0; r < 8; ++r) {
      int m = row0 + r;
      int idx = m * 64 + n;
      float d = acc[q][r] + bv;
      d = d > 0.0f ? d : 0.01f * d;   // leaky_relu
      float u = ug[idx] + d;
      ug[idx] = u;
      ugb[m * 128 + n] = (bf16_t)u;   // refresh A buffer's ugb columns
    }
  }
  SCHED_FENCE();
}

// =====================================================================
// Main persistent-scan kernel: 1 wave == 1 workgroup == 16 batch rows
// for all 12 draws. Occupancy is LDS-bound, so allow the full VGPR file.
// =====================================================================
__global__ __launch_bounds__(32)
__attribute__((amdgpu_waves_per_eu(1))) void gern_main(
    const int* __restrict__ ndraw_p, const float* __restrict__ cnd,
    const float* __restrict__ qj, const float* __restrict__ qd,
    const float* __restrict__ qv, const float* __restrict__ hi_j0,
    const float* __restrict__ ci_j0, const float* __restrict__ hi_d0,
    const float* __restrict__ ci_d0, const float* __restrict__ hg_j0,
    const float* __restrict__ cg_j0, const float* __restrict__ hg_d0,
    const float* __restrict__ cg_d0, const float* __restrict__ ug_j0,
    const float* __restrict__ ug_d0, const float* __restrict__ b_ij,
    const float* __restrict__ b_id, const float* __restrict__ b_gj,
    const float* __restrict__ b_gd, const float* __restrict__ b_poj,
    const float* __restrict__ b_pod, const float* __restrict__ b_prj,
    const float* __restrict__ b_prd, const float* __restrict__ b_dj,
    const float* __restrict__ b_dd, const bf16_t* __restrict__ wt,
    float* __restrict__ out) {
  extern __shared__ unsigned char smem[];
  const int lane = threadIdx.x & 31;
  const int rowBase = blockIdx.x * TROWS;

  bf16_t* aIJ = (bf16_t*)(smem + OF_AIJ);
  bf16_t* aID = (bf16_t*)(smem + OF_AID);
  bf16_t* aGJ = (bf16_t*)(smem + OF_AGJ);
  bf16_t* aGD = (bf16_t*)(smem + OF_AGD);
  bf16_t* aDJ = (bf16_t*)(smem + OF_ADJ);
  bf16_t* aDD = (bf16_t*)(smem + OF_ADD);
  float*  sCij = (float*)(smem + OF_CIJ);
  float*  sCid = (float*)(smem + OF_CID);
  float*  sCgj = (float*)(smem + OF_CGJ);
  float*  sCgd = (float*)(smem + OF_CGD);
  float*  sUgj = (float*)(smem + OF_UGJ);
  float*  sUgd = (float*)(smem + OF_UGD);

  // ---- stage static inputs + initial state into the A buffers ----
#pragma clang loop unroll(disable)
  for (int e = 0; e < 128; ++e) {   // cnd 16x256 -> 4 buffers
    int idx = lane + (e << 5);
    int m = idx >> 8, k = idx & 255;
    bf16_t v = (bf16_t)cnd[(size_t)(rowBase + m) * NRDIM + k];
    aIJ[m * 448 + k] = v;
    aID[m * 448 + k] = v;
    aGJ[m * 384 + k] = v;
    aGD[m * 416 + k] = v;
  }
#pragma clang loop unroll(disable)
  for (int e = 0; e < 32; ++e) {    // 16x64 pieces
    int idx = lane + (e << 5);
    int m = idx >> 6, n = idx & 63;
    size_t g = (size_t)(rowBase + m) * NHDIM + n;
    aIJ[m * 448 + 256 + n] = (bf16_t)qj[(size_t)(rowBase + m) * NQDIM + n];
    aID[m * 448 + 256 + n] = (bf16_t)qd[(size_t)(rowBase + m) * NQDIM + n];
    bf16_t hgj = (bf16_t)hg_j0[g];
    aIJ[m * 448 + 320 + n] = hgj;
    aGJ[m * 384 + 320 + n] = hgj;
    aDJ[m * 128 + 64 + n] = hgj;
    bf16_t hgd = (bf16_t)hg_d0[g];
    aID[m * 448 + 320 + n] = hgd;
    aGD[m * 416 + 352 + n] = hgd;
    aDD[m * 128 + 64 + n] = hgd;
    aIJ[m * 448 + 384 + n] = (bf16_t)hi_j0[g];
    aID[m * 448 + 384 + n] = (bf16_t)hi_d0[g];
    aDJ[m * 128 + n] = (bf16_t)ug_j0[g];
    aDD[m * 128 + n] = (bf16_t)ug_d0[g];
    sCij[idx] = ci_j0[g];
    sCid[idx] = ci_d0[g];
    sCgj[idx] = cg_j0[g];
    sCgd[idx] = cg_d0[g];
    sUgj[idx] = ug_j0[g];
    sUgd[idx] = ug_d0[g];
  }
#pragma clang loop unroll(disable)
  for (int e = 0; e < 16; ++e) {    // qry_v padded chunk in A_gd cols 320..351
    int idx = lane + (e << 5);
    int m = idx >> 5, k = idx & 31;
    aGD[m * 416 + 320 + k] =
        (bf16_t)(k < 4 ? qv[(size_t)(rowBase + m) * 4 + k] : 0.0f);
  }

  const int nd = *ndraw_p;
  const size_t slab = (size_t)BATCH * NHDIM;

  for (int t = 0; t < nd; ++t) {
    // ---- priors from current generator states (h slices inside A_ij/A_id) --
    head_fused(aIJ + 320, 448, wt + WS_PRJ, b_prj,
               out + (size_t)(2 + 0 * nd + t) * slab,
               out + (size_t)(2 + 1 * nd + t) * slab, nullptr, 0, 0, 0u, 0u,
               rowBase, lane);
    head_fused(aID + 320, 448, wt + WS_PRD, b_prd,
               out + (size_t)(2 + 2 * nd + t) * slab,
               out + (size_t)(2 + 3 * nd + t) * slab, nullptr, 0, 0, 0u, 0u,
               rowBase, lane);

    // ---- inference LSTMs ----
    {
      HDst hd[3] = {{aIJ, 448, 384}, {nullptr, 0, 0}, {nullptr, 0, 0}};
      lstm_fused<14>(aIJ, wt + WS_IJ, b_ij, sCij, hd, 0.3f, lane);
    }
    {
      HDst hd[3] = {{aID, 448, 384}, {nullptr, 0, 0}, {nullptr, 0, 0}};
      lstm_fused<14>(aID, wt + WS_ID, b_id, sCid, hd, 0.2f, lane);
    }

    // ---- posteriors + reparameterized samples (z -> generator A buffers) ---
    head_fused(aIJ + 384, 448, wt + WS_POJ, b_poj,
               out + (size_t)(2 + 4 * nd + t) * slab,
               out + (size_t)(2 + 5 * nd + t) * slab, aGJ, 384, 256, 0u,
               (unsigned)t, rowBase, lane);
    head_fused(aID + 384, 448, wt + WS_POD, b_pod,
               out + (size_t)(2 + 6 * nd + t) * slab,
               out + (size_t)(2 + 7 * nd + t) * slab, aGD, 416, 256, 1u,
               (unsigned)t, rowBase, lane);

    // ---- generator LSTMs (new h broadcast to all consumers) ----
    {
      HDst hd[3] = {{aIJ, 448, 320}, {aGJ, 384, 320}, {aDJ, 128, 64}};
      lstm_fused<12>(aGJ, wt + WS_GJ, b_gj, sCgj, hd, 0.3f, lane);
    }
    {
      HDst hd[3] = {{aID, 448, 320}, {aGD, 416, 352}, {aDD, 128, 64}};
      lstm_fused<13>(aGD, wt + WS_GD, b_gd, sCgd, hd, 0.2f, lane);
    }

    // ---- additive canvas updates: ug += leaky_relu([ug|hg]W + b) ----
    delta_fused(aDJ, wt + WS_DJ, b_dj, sUgj, aDJ, lane);
    delta_fused(aDD, wt + WS_DD, b_dd, sUgd, aDD, lane);
  }

  // ---- final canvases ----
#pragma clang loop unroll(disable)
  for (int e = 0; e < 32; ++e) {
    int idx = lane + (e << 5);
    size_t g = (size_t)rowBase * NHDIM + idx;
    out[g] = sUgj[idx];
    out[slab + g] = sUgd[idx];
  }
}

// =====================================================================
extern "C" void kernel_launch(void* const* d_in, const int* in_sizes, int n_in,
                              void* d_out, int out_size, void* d_ws,
                              size_t ws_size, hipStream_t stream) {
  (void)in_sizes; (void)n_in; (void)out_size; (void)ws_size;
  bf16_t* ws = (bf16_t*)d_ws;

  struct PrepArg { int src; int fanIn, fanOut, numKt, numNt; unsigned ofs; int remap; };
  const PrepArg preps[10] = {
      {19, 448, 256, 14, 16, WS_IJ, 0},  {21, 448, 256, 14, 16, WS_ID, 0},
      {23, 384, 256, 12, 16, WS_GJ, 0},  {25, 388, 256, 13, 16, WS_GD, 1},
      {27, 64, 128, 2, 8, WS_POJ, 0},    {29, 64, 128, 2, 8, WS_POD, 0},
      {31, 64, 128, 2, 8, WS_PRJ, 0},    {33, 64, 128, 2, 8, WS_PRD, 0},
      {35, 128, 64, 4, 4, WS_DJ, 0},     {37, 128, 64, 4, 4, WS_DD, 0}};
  for (int i = 0; i < 10; ++i) {
    const PrepArg& p = preps[i];
    int total = p.numKt * p.numNt * 512;
    gern_prep_w<<<(total + 255) / 256, 256, 0, stream>>>(
        (const float*)d_in[p.src], p.fanIn, p.fanOut, p.numKt, p.numNt,
        ws + p.ofs, p.remap);
  }

  dim3 grid(BATCH / TROWS);   // 512 single-wave workgroups
  dim3 block(32);
  size_t lds = PER_WAVE;
  gern_main<<<grid, block, lds, stream>>>(
      (const int*)d_in[0], (const float*)d_in[1], (const float*)d_in[2],
      (const float*)d_in[3], (const float*)d_in[4], (const float*)d_in[5],
      (const float*)d_in[6], (const float*)d_in[8], (const float*)d_in[9],
      (const float*)d_in[11], (const float*)d_in[12], (const float*)d_in[14],
      (const float*)d_in[15], (const float*)d_in[17], (const float*)d_in[18],
      (const float*)d_in[20], (const float*)d_in[22], (const float*)d_in[24],
      (const float*)d_in[26], (const float*)d_in[28], (const float*)d_in[30],
      (const float*)d_in[32], (const float*)d_in[34], (const float*)d_in[36],
      (const float*)d_in[38], ws, (float*)d_out);
}
